// AttentionGateMamba_65360812311161
// MI455X (gfx1250) — matrix operand order repaired and checked by reading, not compiled, not run on hardware
//
#include <hip/hip_runtime.h>
#include <hip/hip_bf16.h>
#include <math.h>
#include <stdint.h>

// ---------------- problem constants ----------------
#define BB     4
#define CE     64
#define CD     128
#define CI     32
#define HH     160
#define WW     160
#define HD     80
#define WD     80
#define LSEQ   (HH*WW)          // 25600
#define PTOT   (BB*LSEQ)        // 102400
#define DI     48               // D_INNER
#define DS     8                // D_STATE
#define DTR    2                // DT_RANK
#define KCV    4                // K_CONV
#define CHUNK  256
#define NCH    (LSEQ/CHUNK)     // 100
#define EPSBN  1e-5f

typedef __attribute__((ext_vector_type(16))) _Float16 v16h;
typedef __attribute__((ext_vector_type(8)))  float    v8f;
typedef __attribute__((ext_vector_type(4)))  unsigned v4u;
typedef __attribute__((ext_vector_type(8)))  unsigned v8u;

__device__ __forceinline__ v8f wmma_f16(v16h a, v16h b, v8f c) {
  // D = A(16x32 f16) * B(32x16 f16) + C(16x16 f32)
  return __builtin_amdgcn_wmma_f32_16x16x32_f16(
      /*neg_a=*/false, a, /*neg_b=*/false, b,
      /*c_mod=*/(short)0, c, /*reuse_a=*/false, /*reuse_b=*/false);
}

// A-fragment half-slot j (0..15) -> K index (within 32-wide k-tile)
__device__ __forceinline__ int a_k(int lane, int j) {
  int kbase = (lane >> 4) * 16;
  return kbase + j;
}

// ----------------------------------------------------------------------------
// Tensor Data Mover: DMA a 2-D f32 tile (rows x tile_w elems, row stride in
// elems) from global memory into LDS.  Wave-uniform SGPR descriptor, one
// tensor_load_to_lds per wave; tracked by TENSORcnt.
// D# layout per CDNA5 ISA ch.8 (group0 128b, group1 256b; VADDR2/3 = NULL
// for 2-D tensors).
// ----------------------------------------------------------------------------
__device__ __forceinline__ void tdm_load_2d_f32(
    unsigned lds_addr, const float* gaddr,
    unsigned rows, unsigned row_stride_elems, unsigned tile_w, unsigned tensor_w)
{
  unsigned long long ga = (unsigned long long)(size_t)gaddr;
  v4u g0;
  g0[0] = 1u;                                            // count=1 (valid D#)
  g0[1] = lds_addr;                                      // LDS byte address
  g0[2] = (unsigned)(ga & 0xFFFFFFFFu);                  // global_addr[31:0]
  g0[3] = (unsigned)((ga >> 32) & 0x1FFFFFFu)            // global_addr[56:32]
        | (2u << 30);                                    // type = 2 ("image")
  v8u g1;
  g1[0] = (2u << 16);                                    // data_size=4B, mask=0
  g1[1] = (tensor_w & 0xFFFFu) << 16;                    // tensor_dim0 lo16
  g1[2] = (tensor_w >> 16) | ((rows & 0xFFFFu) << 16);   // dim0 hi | dim1 lo
  g1[3] = (rows >> 16) | ((tile_w & 0xFFFFu) << 16);     // dim1 hi | tile_dim0
  g1[4] = rows & 0xFFFFu;                                // tile_dim1 (tile_dim2=0)
  g1[5] = row_stride_elems;                              // tensor_dim0_stride lo32
  g1[6] = 0u;                                            // stride0 hi | stride1 lo
  g1[7] = 0u;                                            // stride1 hi
  asm volatile("tensor_load_to_lds %0, %1" :: "s"(g0), "s"(g1) : "memory");
}

// LDS byte offset of a __shared__ object: generic-pointer low 32 bits
// (LDS aperture lives in addr[63:32]; addr[31:0] is the LDS offset).
__device__ __forceinline__ unsigned lds_off(const void* p) {
  return (unsigned)(size_t)p;
}

// ============================================================================
// K1: psi = relu( BN(conv1x1(resize(dec),Wg)) + BN(conv1x1(enc,Wx)) )  -> f16
// Block: 256 thr (8 waves), 128 pixels of one batch.  grid = PTOT/128
// Encoder tile staged by TDM (f32); resized decoder tile computed in VALU.
// ============================================================================
__global__ void __launch_bounds__(256) k_gate_psi(
    const float* __restrict__ enc, const float* __restrict__ dec,
    const float* __restrict__ Wg, const float* __restrict__ g_gamma, const float* __restrict__ g_beta,
    const float* __restrict__ Wx, const float* __restrict__ x_gamma, const float* __restrict__ x_beta,
    _Float16* __restrict__ psi_h)
{
  __shared__ float    s_encf[CE * 128];   // [c][p]  raw f32, TDM destination
  __shared__ _Float16 s_dec[CD * 128];    // [c][p]  bilinear-resized, f16
  const int tid   = threadIdx.x;
  const int blk   = blockIdx.x;
  const int b     = blk / (LSEQ / 128);
  const int pbase = (blk % (LSEQ / 128)) * 128;
  const int lane  = tid & 31, wave = tid >> 5;

  // --- TDM: wave w DMAs encoder channels [w*8, w*8+8) x 128 px into LDS ---
  {
    unsigned c0 = (unsigned)wave * 8u;
    tdm_load_2d_f32(lds_off(&s_encf[c0 * 128]),
                    enc + ((size_t)(b * CE + c0)) * LSEQ + pbase,
                    /*rows=*/8u, /*row_stride=*/LSEQ, /*tile_w=*/128u,
                    /*tensor_w=*/LSEQ);
  }

  // --- overlap: bilinear-resized decoder tile (align_corners mapping) ---
  for (int idx = tid; idx < CD * 128; idx += 256) {
    int c = idx >> 7, p = idx & 127;
    int pp = pbase + p;
    int hh = pp / WW, ww = pp % WW;
    float fy = (float)hh * ((float)(HD - 1) / (float)(HH - 1));
    float fx = (float)ww * ((float)(WD - 1) / (float)(WW - 1));
    int y0 = (int)fy, x0 = (int)fx;
    int y1 = (y0 + 1 > HD - 1) ? (HD - 1) : y0 + 1;
    int x1 = (x0 + 1 > WD - 1) ? (WD - 1) : x0 + 1;
    float wy = fy - (float)y0, wx = fx - (float)x0;
    const float* bp = dec + ((size_t)(b * CD + c)) * (HD * WD);
    float v00 = bp[y0 * WD + x0], v01 = bp[y0 * WD + x1];
    float v10 = bp[y1 * WD + x0], v11 = bp[y1 * WD + x1];
    float top = v00 + (v01 - v00) * wx;
    float bot = v10 + (v11 - v10) * wx;
    s_dec[idx] = (_Float16)(top + (bot - top) * wy);
  }

  __builtin_amdgcn_s_wait_tensorcnt(0);   // this wave's TDM done
  __syncthreads();                        // all waves' slabs + dec tile visible

  const int px0 = wave * 16;
  const int m = lane & 15, nlane = lane & 15, kgrp = (lane >> 4) * 16;

  for (int nt = 0; nt < 2; ++nt) {           // Ci=32 -> 2 N-tiles
    const int o = nt * 16 + nlane;
    const float sg = g_gamma[o] * rsqrtf(1.0f + EPSBN);
    const float sx = x_gamma[o] * rsqrtf(1.0f + EPSBN);
    v8f acc = {};
    // Wg path: K = 128 (4 k-tiles)
    for (int kt = 0; kt < 4; ++kt) {
      v16h af, bf;
      #pragma unroll
      for (int j = 0; j < 16; ++j) {
        af[j] = s_dec[(kt * 32 + a_k(lane, j)) * 128 + px0 + m];
        bf[j] = (_Float16)(Wg[o * CD + kt * 32 + kgrp + j] * sg);
      }
      acc = wmma_f16(af, bf, acc);
    }
    // Wx path: K = 64 (2 k-tiles)
    for (int kt = 0; kt < 2; ++kt) {
      v16h af, bf;
      #pragma unroll
      for (int j = 0; j < 16; ++j) {
        af[j] = (_Float16)s_encf[(kt * 32 + a_k(lane, j)) * 128 + px0 + m];
        bf[j] = (_Float16)(Wx[o * CE + kt * 32 + kgrp + j] * sx);
      }
      acc = wmma_f16(af, bf, acc);
    }
    #pragma unroll
    for (int r = 0; r < 8; ++r) {
      int px = r + (lane >> 4) * 8;
      int ch = nt * 16 + nlane;
      float v = acc[r] + g_beta[ch] + x_beta[ch];
      v = v > 0.0f ? v : 0.0f;                     // ReLU
      psi_h[((size_t)b * LSEQ + pbase + px0 + px) * CI + ch] = (_Float16)v;
    }
  }
}

// ============================================================================
// K2: xz = psi @ in_proj_w^T  (32 -> 96), split xm / z.  grid = PTOT/128
// ============================================================================
__global__ void __launch_bounds__(256) k_inproj(
    const _Float16* __restrict__ psi_h, const float* __restrict__ Wi,
    float* __restrict__ xm, float* __restrict__ zb)
{
  int tid = threadIdx.x, lane = tid & 31, wave = tid >> 5;
  size_t p0 = (size_t)blockIdx.x * 128 + wave * 16;   // global pixel index
  int m = lane & 15, nlane = lane & 15, kgrp = (lane >> 4) * 16;
  v16h af;
  #pragma unroll
  for (int j = 0; j < 16; ++j)
    af[j] = psi_h[(p0 + m) * CI + a_k(lane, j)];
  for (int nt = 0; nt < 6; ++nt) {            // 96 outputs -> 6 N-tiles
    int o = nt * 16 + nlane;
    v16h bf;
    #pragma unroll
    for (int j = 0; j < 16; ++j) bf[j] = (_Float16)Wi[o * CI + kgrp + j];
    v8f acc = {};
    acc = wmma_f16(af, bf, acc);
    #pragma unroll
    for (int r = 0; r < 8; ++r) {
      size_t p = p0 + r + (lane >> 4) * 8;
      int oo = nt * 16 + nlane;
      if (oo < DI) xm[p * DI + oo] = acc[r];
      else         zb[p * DI + (oo - DI)] = acc[r];
    }
  }
}

// ============================================================================
// K3: depthwise causal conv(K=4)+SiLU -> xc;  x_proj 48->18;  dt=softplus(..)
// one thread per position.  grid = PTOT/256
// ============================================================================
__global__ void __launch_bounds__(256) k_convdt(
    const float* __restrict__ xm, const float* __restrict__ conv_w, const float* __restrict__ conv_b,
    const float* __restrict__ xproj_w, const float* __restrict__ dtproj_w, const float* __restrict__ dtproj_b,
    float* __restrict__ xc, float* __restrict__ dtb,
    float* __restrict__ Bmb, float* __restrict__ Cmb)
{
  size_t g = (size_t)blockIdx.x * 256 + threadIdx.x;   // 0..PTOT-1
  int t = (int)(g % LSEQ);                             // position within batch seq
  float xcv[DI];
  #pragma unroll
  for (int d = 0; d < DI; ++d) {
    float a = conv_b[d];
    #pragma unroll
    for (int k = 0; k < KCV; ++k) {
      int tt = t - (KCV - 1) + k;
      if (tt >= 0) a += conv_w[d * KCV + k] * xm[(g - (KCV - 1) + k) * DI + d];
    }
    float sg = 1.0f / (1.0f + __expf(-a));
    xcv[d] = a * sg;                                   // SiLU
    xc[g * DI + d] = xcv[d];
  }
  float dbl[DTR + 2 * DS];
  for (int e = 0; e < DTR + 2 * DS; ++e) {
    float s = 0.0f;
    #pragma unroll
    for (int d = 0; d < DI; ++d) s += xproj_w[e * DI + d] * xcv[d];
    dbl[e] = s;
  }
  #pragma unroll
  for (int d = 0; d < DI; ++d) {
    float s = dtproj_b[d] + dtproj_w[d * 2] * dbl[0] + dtproj_w[d * 2 + 1] * dbl[1];
    dtb[g * DI + d] = (s > 20.0f) ? s : log1pf(__expf(s));   // softplus
  }
  #pragma unroll
  for (int n = 0; n < DS; ++n) {
    Bmb[g * DS + n] = dbl[DTR + n];
    Cmb[g * DS + n] = dbl[DTR + DS + n];
  }
}

// ============================================================================
// K4-K6: chunked associative scan; 384 threads = one lane per (d,n) state
// ============================================================================
__global__ void __launch_bounds__(384) k_scan_phase1(
    const float* __restrict__ dtb, const float* __restrict__ xc, const float* __restrict__ Bmb,
    const float* __restrict__ A_log,
    float* __restrict__ carryA, float* __restrict__ carryB)
{
  int tid = threadIdx.x;
  int d = tid >> 3, n = tid & 7;
  int b = blockIdx.x / NCH, ch = blockIdx.x % NCH;
  float A = -__expf(A_log[d * DS + n]);
  float aacc = 1.0f, bacc = 0.0f;
  size_t base = (size_t)b * LSEQ + (size_t)ch * CHUNK;
  for (int tt = 0; tt < CHUNK; ++tt) {
    size_t g = base + tt;
    float dtv = dtb[g * DI + d];
    float dA  = __expf(dtv * A);
    float bt  = dtv * xc[g * DI + d] * Bmb[g * DS + n];
    bacc = dA * bacc + bt;
    aacc *= dA;
  }
  size_t ci = (size_t)blockIdx.x * 384 + tid;
  carryA[ci] = aacc; carryB[ci] = bacc;
}

__global__ void __launch_bounds__(384) k_scan_phase2(
    const float* __restrict__ carryA, const float* __restrict__ carryB,
    float* __restrict__ carryIn)
{
  int tid = threadIdx.x, b = blockIdx.x;
  float h = 0.0f;
  for (int ch = 0; ch < NCH; ++ch) {
    size_t ci = (size_t)(b * NCH + ch) * 384 + tid;
    carryIn[ci] = h;
    h = carryA[ci] * h + carryB[ci];
  }
}

__global__ void __launch_bounds__(384) k_scan_phase3(
    const float* __restrict__ dtb, const float* __restrict__ xc, const float* __restrict__ Bmb,
    const float* __restrict__ Cmb, const float* __restrict__ zb,
    const float* __restrict__ A_log, const float* __restrict__ D_skip,
    const float* __restrict__ carryIn, _Float16* __restrict__ ygated)
{
  int tid = threadIdx.x;
  int d = tid >> 3, n = tid & 7;
  int b = blockIdx.x / NCH, ch = blockIdx.x % NCH;
  float A = -__expf(A_log[d * DS + n]);
  float h = carryIn[(size_t)blockIdx.x * 384 + tid];
  float Dd = D_skip[d];
  size_t base = (size_t)b * LSEQ + (size_t)ch * CHUNK;
  for (int tt = 0; tt < CHUNK; ++tt) {
    size_t g = base + tt;
    float dtv = dtb[g * DI + d];
    float xcv = xc[g * DI + d];
    float dA  = __expf(dtv * A);
    h = dA * h + dtv * xcv * Bmb[g * DS + n];
    float part = h * Cmb[g * DS + n];
    // reduce over n (8 contiguous lanes within the wave32)
    part += __shfl_xor(part, 4, 32);
    part += __shfl_xor(part, 2, 32);
    part += __shfl_xor(part, 1, 32);
    if (n == 0) {
      float y  = part + xcv * Dd;
      float zv = zb[g * DI + d];
      y *= zv / (1.0f + __expf(-zv));            // y * silu(z)
      ygated[g * DI + d] = (_Float16)y;
    }
  }
}

// ============================================================================
// K7: mamba_out = ygated @ out_proj_w^T  (48 -> 32, K padded to 64)
// ============================================================================
__global__ void __launch_bounds__(256) k_outproj(
    const _Float16* __restrict__ ygated, const float* __restrict__ Wo,
    float* __restrict__ mo)
{
  int tid = threadIdx.x, lane = tid & 31, wave = tid >> 5;
  size_t p0 = (size_t)blockIdx.x * 128 + wave * 16;
  int m = lane & 15, nlane = lane & 15, kgrp = (lane >> 4) * 16;
  v16h a0, a1;
  #pragma unroll
  for (int j = 0; j < 16; ++j) {
    int k = a_k(lane, j);
    a0[j] = ygated[(p0 + m) * DI + k];
    int k1 = 32 + k;
    a1[j] = (k1 < DI) ? ygated[(p0 + m) * DI + k1] : (_Float16)0.0f;
  }
  for (int nt = 0; nt < 2; ++nt) {
    int o = nt * 16 + nlane;
    v16h b0, b1;
    #pragma unroll
    for (int j = 0; j < 16; ++j) {
      b0[j] = (_Float16)Wo[o * DI + kgrp + j];
      int k1 = 32 + kgrp + j;
      b1[j] = (k1 < DI) ? (_Float16)Wo[o * DI + k1] : (_Float16)0.0f;
    }
    v8f acc = {};
    acc = wmma_f16(a0, b0, acc);
    acc = wmma_f16(a1, b1, acc);
    #pragma unroll
    for (int r = 0; r < 8; ++r) {
      size_t p = p0 + r + (lane >> 4) * 8;
      mo[p * CI + nt * 16 + nlane] = acc[r];
    }
  }
}

// ============================================================================
// K8: alpha = sigmoid(BN(psi_w . mamba_out)); out = BN(conv1x1(enc*alpha)+b)
// Encoder tile staged by TDM (raw f32); alpha folded into A-fragment build
// (the A-fragment pixel is constant per lane).
// ============================================================================
__global__ void __launch_bounds__(256) k_final(
    const float* __restrict__ enc, const float* __restrict__ mo,
    const float* __restrict__ psi_w, const float* __restrict__ psi_gamma, const float* __restrict__ psi_beta,
    const float* __restrict__ out_w, const float* __restrict__ out_b,
    const float* __restrict__ out_gamma, const float* __restrict__ out_beta,
    float* __restrict__ out)
{
  __shared__ float s_encf[CE * 128];   // raw encoder tile, TDM destination
  __shared__ float s_alpha[128];
  int tid   = threadIdx.x;
  int blk   = blockIdx.x;
  int b     = blk / (LSEQ / 128);
  int pbase = (blk % (LSEQ / 128)) * 128;
  int lane  = tid & 31, wave = tid >> 5;

  // --- TDM: wave w DMAs encoder channels [w*8, w*8+8) x 128 px into LDS ---
  {
    unsigned c0 = (unsigned)wave * 8u;
    tdm_load_2d_f32(lds_off(&s_encf[c0 * 128]),
                    enc + ((size_t)(b * CE + c0)) * LSEQ + pbase,
                    /*rows=*/8u, /*row_stride=*/LSEQ, /*tile_w=*/128u,
                    /*tensor_w=*/LSEQ);
  }

  // --- overlap: attention coefficient per pixel ---
  if (tid < 128) {
    size_t p = (size_t)b * LSEQ + pbase + tid;
    float s = 0.0f;
    #pragma unroll
    for (int c = 0; c < CI; ++c) s += psi_w[c] * mo[p * CI + c];
    s = s * (psi_gamma[0] * rsqrtf(1.0f + EPSBN)) + psi_beta[0];
    s_alpha[tid] = 1.0f / (1.0f + __expf(-s));
  }

  __builtin_amdgcn_s_wait_tensorcnt(0);
  __syncthreads();

  int px0 = wave * 16;
  int m = lane & 15, nlane = lane & 15, kgrp = (lane >> 4) * 16;
  float aval = s_alpha[px0 + m];      // this lane's pixel gate
  for (int nt = 0; nt < 4; ++nt) {                 // 64 outputs -> 4 N-tiles
    int o = nt * 16 + nlane;
    float so = out_gamma[o] * rsqrtf(1.0f + EPSBN);
    v8f acc = {};
    for (int kt = 0; kt < 2; ++kt) {               // K=64 -> 2 k-tiles
      v16h af, bf;
      #pragma unroll
      for (int j = 0; j < 16; ++j) {
        af[j] = (_Float16)(s_encf[(kt * 32 + a_k(lane, j)) * 128 + px0 + m] * aval);
        bf[j] = (_Float16)(out_w[o * CE + kt * 32 + kgrp + j] * so);
      }
      acc = wmma_f16(af, bf, acc);
    }
    float bias = out_b[o] * so + out_beta[o];
    #pragma unroll
    for (int r = 0; r < 8; ++r) {
      int px = r + (lane >> 4) * 8;
      int chn = nt * 16 + nlane;
      out[((size_t)(b * CE + chn)) * LSEQ + pbase + px0 + px] = acc[r] + bias;
    }
  }
}

// ============================================================================
extern "C" void kernel_launch(void* const* d_in, const int* in_sizes, int n_in,
                              void* d_out, int out_size, void* d_ws, size_t ws_size,
                              hipStream_t stream) {
  (void)in_sizes; (void)n_in; (void)out_size; (void)ws_size;
  const float* enc       = (const float*)d_in[0];
  const float* dec       = (const float*)d_in[1];
  const float* Wg        = (const float*)d_in[2];
  const float* g_gamma   = (const float*)d_in[3];
  const float* g_beta    = (const float*)d_in[4];
  const float* Wx        = (const float*)d_in[5];
  const float* x_gamma   = (const float*)d_in[6];
  const float* x_beta    = (const float*)d_in[7];
  const float* in_proj_w = (const float*)d_in[8];
  const float* conv_w    = (const float*)d_in[9];
  const float* conv_b    = (const float*)d_in[10];
  const float* xproj_w   = (const float*)d_in[11];
  const float* dtproj_w  = (const float*)d_in[12];
  const float* dtproj_b  = (const float*)d_in[13];
  const float* A_log     = (const float*)d_in[14];
  const float* D_skip    = (const float*)d_in[15];
  const float* out_proj  = (const float*)d_in[16];
  const float* psi_w     = (const float*)d_in[17];
  const float* psi_gam   = (const float*)d_in[18];
  const float* psi_bet   = (const float*)d_in[19];
  const float* out_w     = (const float*)d_in[20];
  const float* out_b     = (const float*)d_in[21];
  const float* out_gam   = (const float*)d_in[22];
  const float* out_bet   = (const float*)d_in[23];

  // ---- workspace carve-up (256B aligned) ----
  char* ws = (char*)d_ws;
  size_t off = 0;
  auto carve = [&](size_t bytes) { void* p = ws + off;
      off += (bytes + 255) & ~(size_t)255; return p; };
  _Float16* psi_h  = (_Float16*)carve((size_t)PTOT * CI * sizeof(_Float16));
  float*    xm     = (float*)   carve((size_t)PTOT * DI * sizeof(float));
  float*    zb     = (float*)   carve((size_t)PTOT * DI * sizeof(float));
  float*    xc     = (float*)   carve((size_t)PTOT * DI * sizeof(float));
  float*    dtb    = (float*)   carve((size_t)PTOT * DI * sizeof(float));
  float*    Bmb    = (float*)   carve((size_t)PTOT * DS * sizeof(float));
  float*    Cmb    = (float*)   carve((size_t)PTOT * DS * sizeof(float));
  float*    carryA = (float*)   carve((size_t)BB * NCH * DI * DS * sizeof(float));
  float*    carryB = (float*)   carve((size_t)BB * NCH * DI * DS * sizeof(float));
  float*    carryI = (float*)   carve((size_t)BB * NCH * DI * DS * sizeof(float));
  _Float16* ygated = (_Float16*)carve((size_t)PTOT * DI * sizeof(_Float16));
  float*    mo     = (float*)   carve((size_t)PTOT * CI * sizeof(float));

  k_gate_psi<<<PTOT / 128, 256, 0, stream>>>(enc, dec, Wg, g_gamma, g_beta,
                                             Wx, x_gamma, x_beta, psi_h);
  k_inproj<<<PTOT / 128, 256, 0, stream>>>(psi_h, in_proj_w, xm, zb);
  k_convdt<<<PTOT / 256, 256, 0, stream>>>(xm, conv_w, conv_b, xproj_w,
                                           dtproj_w, dtproj_b, xc, dtb, Bmb, Cmb);
  k_scan_phase1<<<BB * NCH, 384, 0, stream>>>(dtb, xc, Bmb, A_log, carryA, carryB);
  k_scan_phase2<<<BB, 384, 0, stream>>>(carryA, carryB, carryI);
  k_scan_phase3<<<BB * NCH, 384, 0, stream>>>(dtb, xc, Bmb, Cmb, zb, A_log,
                                              D_skip, carryI, ygated);
  k_outproj<<<PTOT / 128, 256, 0, stream>>>(ygated, out_proj, mo);
  k_final<<<PTOT / 128, 256, 0, stream>>>(enc, mo, psi_w, psi_gam, psi_bet,
                                          out_w, out_b, out_gam, out_bet,
                                          (float*)d_out);
}
